// CustomLSTM_41231686041826
// MI455X (gfx1250) — compile-verified, matrix operand order checked by reading, not run on hardware
//
#include <hip/hip_runtime.h>

// ---------------------------------------------------------------------------
// CDNA5 (gfx1250, wave32) persistent LSTM:
//   phase 0: convert x / W / fc_w to bf16, zero h double-buffer + barriers
//   phase 1: persistent kernel, 32 blocks x 256 thr; each wave owns one
//            16x16 (batch x hidden) tile; cell state lives in VGPRs across
//            all 512 timesteps; per-step device barrier via arrive counters.
//            Gate GEMM = v_wmma_f32_16x16x32_bf16, K = 1536 (h|x fused).
//            Gate nonlinearities use gfx1250 hardware v_tanh_f32 / v_exp_f32
//            (branch-free) to keep the serial per-step tail short.
//   phase 2: final FC with the same WMMA tiling.
// ---------------------------------------------------------------------------

typedef __bf16 bf16_t;
typedef __attribute__((ext_vector_type(8)))  __bf16 v8bf;
typedef __attribute__((ext_vector_type(16))) __bf16 v16bf;
typedef __attribute__((ext_vector_type(8)))  float  v8f;

constexpr int Bn = 64;      // batch
constexpr int Tn = 512;     // timesteps
constexpr int In = 512;     // input size
constexpr int Hn = 1024;    // hidden size
constexpr int Sn = 512;     // fc output size
constexpr int KC = Hn + In; // 1536 fused K dimension
constexpr int NBLK = 32;    // persistent grid (4 mtiles x 8 hgroups)

union FragBF { v16bf v; v8bf h8[2]; };

// ---- branch-free transcendentals on the CDNA5 TRANS pipe --------------------
__device__ __forceinline__ float fast_exp2(float x) {
#if __has_builtin(__builtin_amdgcn_exp2f)
    return __builtin_amdgcn_exp2f(x);
#else
    return exp2f(x);
#endif
}
__device__ __forceinline__ float fast_rcp(float x) {
#if __has_builtin(__builtin_amdgcn_rcpf)
    return __builtin_amdgcn_rcpf(x);
#else
    return 1.0f / x;
#endif
}
__device__ __forceinline__ float fast_sigmoid(float x) {
    // 1 / (1 + 2^(-x*log2(e)))  -> v_exp_f32 + v_rcp_f32, no branches
    return fast_rcp(1.0f + fast_exp2(x * -1.442695040888963f));
}
__device__ __forceinline__ float fast_tanh(float x) {
#if __has_builtin(__builtin_amdgcn_tanhf)
    return __builtin_amdgcn_tanhf(x);        // gfx1250 hardware v_tanh_f32
#else
    // tanh(x) = 1 - 2/(2^(2x*log2e) + 1), branch-free fallback
    float e = fast_exp2(x * 2.885390081777927f);
    return 1.0f - 2.0f * fast_rcp(e + 1.0f);
#endif
}

// ---------------------------------------------------------------------------
// Phase 0: conversions + zeroing (grid-stride over all segments)
// ---------------------------------------------------------------------------
__global__ void lstm_init(const float* __restrict__ x,
                          const float* __restrict__ Wf, const float* __restrict__ Wi,
                          const float* __restrict__ Wc, const float* __restrict__ Wo,
                          const float* __restrict__ fcw,
                          bf16_t* __restrict__ xbf, bf16_t* __restrict__ wbf,
                          bf16_t* __restrict__ fcbf, bf16_t* __restrict__ hbuf,
                          unsigned* __restrict__ arrive)
{
    const long long NX = (long long)Bn * Tn * In;   // 16,777,216
    const long long NW = 4LL * Hn * KC;             //  6,291,456
    const long long NF = (long long)Sn * Hn;        //    524,288
    const long long NH = 2LL * Bn * Hn;             //    131,072
    const long long NA = Tn;
    const long long total  = NX + NW + NF + NH + NA;
    const long long stride = (long long)gridDim.x * blockDim.x;
    for (long long i = blockIdx.x * (long long)blockDim.x + threadIdx.x;
         i < total; i += stride) {
        if (i < NX) {
            xbf[i] = (bf16_t)x[i];
        } else if (i < NX + NW) {
            long long j = i - NX;
            int g = (int)(j / ((long long)Hn * KC));
            long long r = j % ((long long)Hn * KC);
            const float* W = (g == 0) ? Wf : (g == 1) ? Wi : (g == 2) ? Wc : Wo;
            wbf[j] = (bf16_t)W[r];
        } else if (i < NX + NW + NF) {
            long long j = i - NX - NW;
            fcbf[j] = (bf16_t)fcw[j];
        } else if (i < NX + NW + NF + NH) {
            hbuf[i - NX - NW - NF] = (bf16_t)0.0f;
        } else {
            arrive[i - NX - NW - NF - NH] = 0u;
        }
    }
}

// ---------------------------------------------------------------------------
// Phase 1: persistent recurrent kernel
// block = (mtile = blockIdx%4, hgroup = blockIdx/4); wave w -> htile hgroup*8+w
// ---------------------------------------------------------------------------
__launch_bounds__(256, 1)
__global__ void lstm_steps(const bf16_t* __restrict__ xbf,   // [B][T][I] bf16
                           const bf16_t* __restrict__ wbf,   // [4][H][KC] bf16
                           const float* __restrict__ bfv, const float* __restrict__ biv,
                           const float* __restrict__ bcv, const float* __restrict__ bov,
                           bf16_t* __restrict__ hbuf,        // [2][B][H] bf16
                           unsigned* __restrict__ arrive)    // [T]
{
    __shared__ __align__(16) bf16_t smem[16 * KC];   // 48 KB: A rows (combined)

    const int tid  = threadIdx.x;
    const int wave = tid >> 5;
    const int lane = tid & 31;
    const int mtile = blockIdx.x & 3;
    const int hgrp  = blockIdx.x >> 2;
    const int htile = hgrp * 8 + wave;
    const int m0 = mtile * 16;
    const int n0 = htile * 16;
    const int col  = lane & 15;     // A row / B column / C column
    const int half = lane >> 4;     // K-split half per 16-bit fragment layout

    const float bias_f = bfv[n0 + col];
    const float bias_i = biv[n0 + col];
    const float bias_c = bcv[n0 + col];
    const float bias_o = bov[n0 + col];

    // B-matrix row base pointers (W[n, :] is contiguous in K)
    const bf16_t* wrow0 = wbf + ((long long)0 * Hn + (n0 + col)) * KC;
    const bf16_t* wrow1 = wbf + ((long long)1 * Hn + (n0 + col)) * KC;
    const bf16_t* wrow2 = wbf + ((long long)2 * Hn + (n0 + col)) * KC;
    const bf16_t* wrow3 = wbf + ((long long)3 * Hn + (n0 + col)) * KC;

    v8f c_state;
    #pragma unroll
    for (int r = 0; r < 8; ++r) c_state[r] = 0.0f;

    for (int t = 0; t < Tn; ++t) {
        // ---- wait for h_{t-1} to be globally visible ----
        if (t > 0) {
            if (tid == 0) {
                while (__hip_atomic_load(&arrive[t - 1], __ATOMIC_ACQUIRE,
                                         __HIP_MEMORY_SCOPE_AGENT) < (unsigned)NBLK)
                    __builtin_amdgcn_s_sleep(1);
            }
            __syncthreads();
            __threadfence();
        }

        const bf16_t* hprev = hbuf + (size_t)(t & 1) * Bn * Hn;

        // ---- stage A = [h_prev | x_t] rows m0..m0+15 into LDS (16B chunks) ----
        for (int c = tid; c < 16 * KC / 8; c += 256) {
            int row = (c * 8) / KC;
            int k   = (c * 8) % KC;
            const uint4* src;
            if (k < Hn)
                src = (const uint4*)(hprev + (size_t)(m0 + row) * Hn + k);
            else
                src = (const uint4*)(xbf + ((size_t)(m0 + row) * Tn + t) * In + (k - Hn));
            *(uint4*)(&smem[row * KC + k]) = *src;
        }
        __syncthreads();

        // ---- 4-gate GEMM, K = 1536, bias folded into C init ----
        v8f accf, acci, accc, acco;
        #pragma unroll
        for (int r = 0; r < 8; ++r) {
            accf[r] = bias_f; acci[r] = bias_i; accc[r] = bias_c; acco[r] = bias_o;
        }

        const bf16_t* arow = &smem[col * KC];
        #pragma unroll 2
        for (int kc = 0; kc < KC; kc += 32) {
            FragBF a;
            a.h8[0] = *(const v8bf*)(arow + kc + 8 * half);
            a.h8[1] = *(const v8bf*)(arow + kc + 16 + 8 * half);
            FragBF b0, b1, b2, b3;
            b0.h8[0] = *(const v8bf*)(wrow0 + kc + 8 * half);
            b0.h8[1] = *(const v8bf*)(wrow0 + kc + 16 + 8 * half);
            b1.h8[0] = *(const v8bf*)(wrow1 + kc + 8 * half);
            b1.h8[1] = *(const v8bf*)(wrow1 + kc + 16 + 8 * half);
            b2.h8[0] = *(const v8bf*)(wrow2 + kc + 8 * half);
            b2.h8[1] = *(const v8bf*)(wrow2 + kc + 16 + 8 * half);
            b3.h8[0] = *(const v8bf*)(wrow3 + kc + 8 * half);
            b3.h8[1] = *(const v8bf*)(wrow3 + kc + 16 + 8 * half);
            accf = __builtin_amdgcn_wmma_f32_16x16x32_bf16(false, a.v, false, b0.v,
                                                           (short)0, accf, false, false);
            acci = __builtin_amdgcn_wmma_f32_16x16x32_bf16(false, a.v, false, b1.v,
                                                           (short)0, acci, false, false);
            accc = __builtin_amdgcn_wmma_f32_16x16x32_bf16(false, a.v, false, b2.v,
                                                           (short)0, accc, false, false);
            acco = __builtin_amdgcn_wmma_f32_16x16x32_bf16(false, a.v, false, b3.v,
                                                           (short)0, acco, false, false);
        }

        // ---- gate nonlinearities + register-resident cell update -------------
        // Branch-free: v_exp_f32 / v_rcp_f32 / v_tanh_f32 on the TRANS pipe.
        bf16_t* hnext = hbuf + (size_t)((t + 1) & 1) * Bn * Hn;
        #pragma unroll
        for (int r = 0; r < 8; ++r) {
            float fg = fast_sigmoid(accf[r]);
            float ig = fast_sigmoid(acci[r]);
            float cg = fast_tanh(accc[r]);
            float og = fast_sigmoid(acco[r]);
            float cn = fg * c_state[r] + ig * cg;
            c_state[r] = cn;
            float hv = og * fast_tanh(cn);
            int m = m0 + r + half * 8;                      // C/D VGPR layout
            hnext[(size_t)m * Hn + n0 + col] = (bf16_t)hv;
        }

        // ---- publish h_t ----
        __threadfence();
        __syncthreads();
        if (tid == 0)
            __hip_atomic_fetch_add(&arrive[t], 1u, __ATOMIC_RELEASE,
                                   __HIP_MEMORY_SCOPE_AGENT);
    }
}

// ---------------------------------------------------------------------------
// Phase 2: out = h_T @ fc_w^T + fc_b   (M=64, N=512, K=1024)
// ---------------------------------------------------------------------------
__launch_bounds__(256, 1)
__global__ void fc_kernel(const bf16_t* __restrict__ hfin,  // [B][H] bf16 (hbuf[0])
                          const bf16_t* __restrict__ fcbf,  // [S][H] bf16
                          const float* __restrict__ fcb,
                          float* __restrict__ out)          // [B][S] f32
{
    const int tid  = threadIdx.x;
    const int wave = tid >> 5;
    const int lane = tid & 31;
    const int tile = blockIdx.x * 8 + wave;  // 0..127
    const int mtile = tile >> 5;             // 0..3
    const int ntile = tile & 31;             // 0..31
    const int m0 = mtile * 16, n0 = ntile * 16;
    const int col = lane & 15, half = lane >> 4;

    v8f acc;
    const float b = fcb[n0 + col];
    #pragma unroll
    for (int r = 0; r < 8; ++r) acc[r] = b;

    const bf16_t* arow = hfin + (size_t)(m0 + col) * Hn;
    const bf16_t* brow = fcbf + (size_t)(n0 + col) * Hn;
    #pragma unroll 4
    for (int kc = 0; kc < Hn; kc += 32) {
        FragBF a, bb;
        a.h8[0]  = *(const v8bf*)(arow + kc + 8 * half);
        a.h8[1]  = *(const v8bf*)(arow + kc + 16 + 8 * half);
        bb.h8[0] = *(const v8bf*)(brow + kc + 8 * half);
        bb.h8[1] = *(const v8bf*)(brow + kc + 16 + 8 * half);
        acc = __builtin_amdgcn_wmma_f32_16x16x32_bf16(false, a.v, false, bb.v,
                                                      (short)0, acc, false, false);
    }
    #pragma unroll
    for (int r = 0; r < 8; ++r) {
        int m = m0 + r + half * 8;
        out[(size_t)m * Sn + n0 + col] = acc[r];
    }
}

// ---------------------------------------------------------------------------
extern "C" void kernel_launch(void* const* d_in, const int* in_sizes, int n_in,
                              void* d_out, int out_size, void* d_ws, size_t ws_size,
                              hipStream_t stream) {
    const float* x   = (const float*)d_in[0];
    const float* Wf  = (const float*)d_in[1];
    const float* bf_ = (const float*)d_in[2];
    const float* Wi  = (const float*)d_in[3];
    const float* bi_ = (const float*)d_in[4];
    const float* Wc  = (const float*)d_in[5];
    const float* bc_ = (const float*)d_in[6];
    const float* Wo  = (const float*)d_in[7];
    const float* bo_ = (const float*)d_in[8];
    const float* fcw = (const float*)d_in[9];
    const float* fcb = (const float*)d_in[10];
    float* out = (float*)d_out;

    char* ws = (char*)d_ws;
    size_t off = 0;
    auto carve = [&](size_t bytes) -> void* {
        void* p = ws + off;
        off = (off + bytes + 255) & ~(size_t)255;
        return p;
    };
    bf16_t*   xbf    = (bf16_t*)  carve((size_t)Bn * Tn * In * sizeof(bf16_t));
    bf16_t*   wbf    = (bf16_t*)  carve((size_t)4 * Hn * KC * sizeof(bf16_t));
    bf16_t*   fcbf   = (bf16_t*)  carve((size_t)Sn * Hn * sizeof(bf16_t));
    bf16_t*   hbuf   = (bf16_t*)  carve((size_t)2 * Bn * Hn * sizeof(bf16_t));
    unsigned* arrive = (unsigned*)carve((size_t)Tn * sizeof(unsigned));
    (void)ws_size; (void)in_sizes; (void)n_in; (void)out_size;

    lstm_init<<<4096, 256, 0, stream>>>(x, Wf, Wi, Wc, Wo, fcw,
                                        xbf, wbf, fcbf, hbuf, arrive);
    lstm_steps<<<NBLK, 256, 0, stream>>>(xbf, wbf, bf_, bi_, bc_, bo_,
                                         hbuf, arrive);
    // T even -> final h lives in hbuf[0]
    fc_kernel<<<16, 256, 0, stream>>>(hbuf, fcbf, fcb, out);
}
